// Attention_14104672600564
// MI455X (gfx1250) — matrix operand order repaired and checked by reading, not compiled, not run on hardware
//
#include <hip/hip_runtime.h>

typedef __attribute__((ext_vector_type(16))) _Float16 v16h;
typedef __attribute__((ext_vector_type(8)))  float    v8f;
typedef unsigned int __attribute__((ext_vector_type(4))) u32x4;
typedef int  __attribute__((ext_vector_type(4))) i32x4;
typedef int  __attribute__((ext_vector_type(8))) i32x8;

#define B_    4
#define C_    128
#define H_    4
#define D_    32
#define NTOK  4096
#define QSCALE 0.17677669529663687f   // 32^-0.5

#if defined(__has_builtin)
#  if __has_builtin(__builtin_amdgcn_tensor_load_to_lds) && \
      __has_builtin(__builtin_amdgcn_s_wait_tensorcnt)
#    define HAVE_TDM 1
#  endif
#endif
#ifndef HAVE_TDM
#  define HAVE_TDM 0
#endif

static __device__ __forceinline__ v8f wmma_f16(v16h a, v16h b, v8f c) {
  // D = A(16x32 f16) * B(32x16 f16) + C(16x16 f32)
  return __builtin_amdgcn_wmma_f32_16x16x32_f16(false, a, false, b, (short)0, c,
                                                false, false);
}

#if HAVE_TDM
// ---------------------------------------------------------------------------
// TDM: build a D# (group0 + group1) for a 2-byte-element 2D tile and issue
// tensor_load_to_lds.  tile_d1==0 -> 1D contiguous copy.
// ---------------------------------------------------------------------------
static __device__ __forceinline__ void tdm_load_tile(
    unsigned lds_addr, const void* gptr,
    unsigned tensor_d0, unsigned tensor_d1,
    unsigned tile_d0, unsigned tile_d1, unsigned stride0) {
  unsigned long long ga = (unsigned long long)gptr;
  u32x4 g0;
  g0[0] = 1u;                                        // count=1 (user mode)
  g0[1] = lds_addr;                                  // lds_addr (bytes)
  g0[2] = (unsigned)(ga & 0xffffffffull);            // global_addr[31:0]
  g0[3] = (unsigned)((ga >> 32) & 0x1ffffffull)      // global_addr[56:32]
        | (2u << 30);                                // type = 2 ("image")
  i32x8 g1;
  g1[0] = (int)(1u << 16);                           // data_size=1 (2 bytes)
  g1[1] = (int)((tensor_d0 & 0xffffu) << 16);        // tensor_dim0[15:0]
  g1[2] = (int)((tensor_d0 >> 16) | ((tensor_d1 & 0xffffu) << 16));
  g1[3] = (int)((tensor_d1 >> 16) | (tile_d0 << 16));// tile_dim0
  g1[4] = (int)(tile_d1 & 0xffffu);                  // tile_dim1, tile_dim2=0
  g1[5] = (int)stride0;                              // tensor_dim0_stride[31:0]
  g1[6] = 0;                                         // stride0[47:32], stride1 lo
  g1[7] = 0;
  i32x4 z4 = {};
#if __clang_major__ >= 23
  i32x8 z8 = {};
  __builtin_amdgcn_tensor_load_to_lds(g0, g1, z4, z4, z8, 0);
#else
  __builtin_amdgcn_tensor_load_to_lds(g0, g1, z4, z4, 0);
#endif
}
static __device__ __forceinline__ unsigned lds_off(const void* p) {
  return (unsigned)(unsigned long long)p;
}
#endif

// ---------------------------------------------------------------------------
// Kernel A: Q/K/V projections. One wave computes a 16(out-ch) x 16(token)
// tile via 4 x v_wmma_f32_16x16x32_f16 over the 128-channel K dim.
// Q,K stored f16 row-major [b][h][token][d]; V stored transposed [b][h][d][token].
// ---------------------------------------------------------------------------
__global__ void proj_qkv(const float* __restrict__ x,
                         const float* __restrict__ wq,
                         const float* __restrict__ wk,
                         const float* __restrict__ wv,
                         _Float16* __restrict__ qws,
                         _Float16* __restrict__ kws,
                         _Float16* __restrict__ vtws) {
  int tile = blockIdx.x;
  int tn = tile & 255;            // token tile (0..255)
  int to = (tile >> 8) & 7;       // out-channel tile (0..7)
  int b  = tile >> 11;            // batch
  int which = blockIdx.y;         // 0=q 1=k 2=v
  const float* w = (which == 0) ? wq : (which == 1) ? wk : wv;

  int lane = threadIdx.x;
  int l15 = lane & 15;
  int hg  = (lane >> 4) & 1;

  const float* xb = x + (size_t)b * C_ * NTOK;

  v8f acc = {};
#pragma unroll
  for (int kb = 0; kb < C_; kb += 32) {
    v16h a;
    {
      const float* p = w + (size_t)(to * 16 + l15) * C_ + kb + hg * 16;
#pragma unroll
      for (int i = 0; i < 8; ++i) {
        a[i]     = (_Float16)p[i];
        a[8 + i] = (_Float16)p[8 + i];
      }
    }
    v16h bf;
    {
      const float* p = xb + (size_t)(kb + hg * 16) * NTOK + tn * 16 + l15;
#pragma unroll
      for (int i = 0; i < 16; ++i) bf[i] = (_Float16)p[(size_t)i * NTOK];
    }
    acc = wmma_f16(a, bf, acc);
  }

  int token = tn * 16 + l15;
#pragma unroll
  for (int r = 0; r < 8; ++r) {
    int o = to * 16 + r + 8 * hg;      // C/D layout: M = r + 8*(lane>=16)
    int h = o >> 5, d = o & 31;
    float v = acc[r];
    if (which == 0) {
      qws[(((size_t)b * H_ + h) * NTOK + token) * D_ + d] =
          (_Float16)(v * QSCALE);
    } else if (which == 1) {
      kws[(((size_t)b * H_ + h) * NTOK + token) * D_ + d] = (_Float16)v;
    } else {
      vtws[(((size_t)b * H_ + h) * D_ + d) * NTOK + token] = (_Float16)v;
    }
  }
}

// ---------------------------------------------------------------------------
// Online-softmax update for one 16x32 S tile held as two C-layout v8f tiles.
// Also writes P (f16) into LDS in row-major [16][32] for the A-frag reload.
// ---------------------------------------------------------------------------
static __device__ __forceinline__ void softmax_update(
    const v8f& S0, const v8f& S1, float* m, float* l, v8f& O0, v8f& O1,
    _Float16* Pt, int l15, int hg) {
#pragma unroll
  for (int r = 0; r < 8; ++r) {
    float s0 = S0[r], s1 = S1[r];
    float bm = fmaxf(s0, s1);
    bm = fmaxf(bm, __shfl_xor(bm, 1, 32));
    bm = fmaxf(bm, __shfl_xor(bm, 2, 32));
    bm = fmaxf(bm, __shfl_xor(bm, 4, 32));
    bm = fmaxf(bm, __shfl_xor(bm, 8, 32));
    float mn    = fmaxf(m[r], bm);
    float alpha = __expf(m[r] - mn);
    float p0    = __expf(s0 - mn);
    float p1    = __expf(s1 - mn);
    float rs = p0 + p1;
    rs += __shfl_xor(rs, 1, 32);
    rs += __shfl_xor(rs, 2, 32);
    rs += __shfl_xor(rs, 4, 32);
    rs += __shfl_xor(rs, 8, 32);
    l[r] = l[r] * alpha + rs;
    m[r] = mn;
    O0[r] *= alpha;
    O1[r] *= alpha;
    int M = r + 8 * hg;
    Pt[M * 32 + l15]      = (_Float16)p0;
    Pt[M * 32 + 16 + l15] = (_Float16)p1;
  }
}

// ---------------------------------------------------------------------------
// Kernel B: flash attention. One wave owns a 32-query tile (two A-frags);
// loops over 4096 keys in blocks of 32. K/V blocks are staged into LDS by the
// Tensor Data Mover (double-buffered, S_WAIT_TENSORCNT pipelined); falls back
// to direct global loads if the TDM builtin is unavailable.
// Per block: 4 S-WMMAs, 2 LDS P-transposes, 4 PV-WMMAs.
// ---------------------------------------------------------------------------
__global__ void flash_attn(const _Float16* __restrict__ qws,
                           const _Float16* __restrict__ kws,
                           const _Float16* __restrict__ vtws,
                           _Float16* __restrict__ ows) {
#if HAVE_TDM
  __shared__ _Float16 Kb[2][32 * 32];   // [key][d]
  __shared__ _Float16 Vb[2][32 * 32];   // [d][key]
#endif
  __shared__ _Float16 Pt[16 * 32];

  int idx = blockIdx.x;
  int qt = idx & 127;                    // 128 query tiles of 32
  int bh = idx >> 7;
  int lane = threadIdx.x;
  int l15 = lane & 15;
  int hg  = (lane >> 4) & 1;

  const _Float16* Qp = qws + ((size_t)bh * NTOK + (size_t)qt * 32) * D_;
  const _Float16* Kp = kws + (size_t)bh * NTOK * D_;
  const _Float16* Vt = vtws + (size_t)bh * D_ * NTOK;

  // Two Q A-fragments (rows f*16 + l15).
  v16h aQ0, aQ1;
  {
    const _Float16* p0 = Qp + l15 * D_ + hg * 16;
    const _Float16* p1 = p0 + 16 * D_;
#pragma unroll
    for (int i = 0; i < 8; ++i) {
      aQ0[i] = p0[i]; aQ0[8 + i] = p0[8 + i];
      aQ1[i] = p1[i]; aQ1[8 + i] = p1[8 + i];
    }
  }

  v8f O00 = {}, O01 = {}, O10 = {}, O11 = {};
  float m0[8], l0[8], m1[8], l1[8];
#pragma unroll
  for (int r = 0; r < 8; ++r) {
    m0[r] = -3.0e38f; l0[r] = 0.f;
    m1[r] = -3.0e38f; l1[r] = 0.f;
  }

#if HAVE_TDM
  // Prologue: fill both buffers (2 tensor ops per block -> TENSORcnt 4).
#pragma unroll
  for (int p = 0; p < 2; ++p) {
    tdm_load_tile(lds_off(&Kb[p][0]), Kp + (size_t)p * 32 * D_,
                  NTOK * D_, 1, 32 * D_, 0, NTOK * D_);        // 1D, 2KB
    tdm_load_tile(lds_off(&Vb[p][0]), Vt + (size_t)p * 32,
                  NTOK, D_, 32, 32, NTOK);                     // 2D 32x32
  }
#endif

  for (int i = 0; i < NTOK / 32; ++i) {
    int j = i * 32;
    v16h bk0, bk1, bv0, bv1;
#if HAVE_TDM
    int buf = i & 1;
    // Block i's pair complete once <=2 tensor ops (block i+1's) remain.
    __builtin_amdgcn_s_wait_tensorcnt(2);
    bk0 = *(const v16h*)&Kb[buf][l15 * 32 + hg * 16];
    bk1 = *(const v16h*)&Kb[buf][(16 + l15) * 32 + hg * 16];
    bv0 = *(const v16h*)&Vb[buf][l15 * 32 + hg * 16];
    bv1 = *(const v16h*)&Vb[buf][(16 + l15) * 32 + hg * 16];
    // Fragments must be in VGPRs before the TDM refills this buffer.
    asm volatile("s_wait_dscnt 0" ::: "memory");
    if (i + 2 < NTOK / 32) {
      int jn = j + 64;
      tdm_load_tile(lds_off(&Kb[buf][0]), Kp + (size_t)jn * D_,
                    NTOK * D_, 1, 32 * D_, 0, NTOK * D_);
      tdm_load_tile(lds_off(&Vb[buf][0]), Vt + (size_t)jn,
                    NTOK, D_, 32, 32, NTOK);
    }
#else
    {
      const _Float16* kp = Kp + (size_t)(j + l15) * D_ + hg * 16;
      bk0 = *(const v16h*)kp;
      bk1 = *(const v16h*)(kp + 16 * D_);
      const _Float16* vp = Vt + (size_t)l15 * NTOK + j + hg * 16;
      bv0 = *(const v16h*)vp;
      bv1 = *(const v16h*)(vp + (size_t)16 * NTOK);
    }
#endif
    v8f zc = {};

    // ---- query fragment 0 ----
    {
      v8f S0 = wmma_f16(aQ0, bk0, zc);
      v8f S1 = wmma_f16(aQ0, bk1, zc);
      softmax_update(S0, S1, m0, l0, O00, O01, Pt, l15, hg);
      asm volatile("s_wait_dscnt 0" ::: "memory");
      v16h aP;
      int base = l15 * 32 + hg * 16;
#pragma unroll
      for (int t = 0; t < 8; ++t) { aP[t] = Pt[base + t]; aP[8 + t] = Pt[base + 8 + t]; }
      O00 = wmma_f16(aP, bv0, O00);
      O01 = wmma_f16(aP, bv1, O01);
    }
    // ---- query fragment 1 ----
    {
      v8f S0 = wmma_f16(aQ1, bk0, zc);
      v8f S1 = wmma_f16(aQ1, bk1, zc);
      softmax_update(S0, S1, m1, l1, O10, O11, Pt, l15, hg);
      asm volatile("s_wait_dscnt 0" ::: "memory");
      v16h aP;
      int base = l15 * 32 + hg * 16;
#pragma unroll
      for (int t = 0; t < 8; ++t) { aP[t] = Pt[base + t]; aP[8 + t] = Pt[base + 8 + t]; }
      O10 = wmma_f16(aP, bv0, O10);
      O11 = wmma_f16(aP, bv1, O11);
    }
  }

  int b = bh >> 2, h = bh & 3;
#pragma unroll
  for (int r = 0; r < 8; ++r) {
    int tok0 = qt * 32 + r + 8 * hg;
    int tok1 = tok0 + 16;
    float i0 = 1.f / l0[r], i1 = 1.f / l1[r];
    ows[((size_t)b * C_ + h * D_ + l15) * NTOK + tok0]      = (_Float16)(O00[r] * i0);
    ows[((size_t)b * C_ + h * D_ + 16 + l15) * NTOK + tok0] = (_Float16)(O01[r] * i0);
    ows[((size_t)b * C_ + h * D_ + l15) * NTOK + tok1]      = (_Float16)(O10[r] * i1);
    ows[((size_t)b * C_ + h * D_ + 16 + l15) * NTOK + tok1] = (_Float16)(O11[r] * i1);
  }
}

// ---------------------------------------------------------------------------
// Kernel C: output projection wo @ O + bo, f32 result.
// ---------------------------------------------------------------------------
__global__ void proj_out(const _Float16* __restrict__ ows,
                         const float* __restrict__ wo,
                         const float* __restrict__ bo,
                         float* __restrict__ out) {
  int tile = blockIdx.x;
  int tn = tile & 255;
  int to = (tile >> 8) & 7;
  int b  = tile >> 11;
  int lane = threadIdx.x;
  int l15 = lane & 15;
  int hg  = (lane >> 4) & 1;

  const _Float16* ob = ows + (size_t)b * C_ * NTOK;
  v8f acc = {};
#pragma unroll
  for (int kb = 0; kb < C_; kb += 32) {
    v16h a;
    {
      const float* p = wo + (size_t)(to * 16 + l15) * C_ + kb + hg * 16;
#pragma unroll
      for (int i = 0; i < 8; ++i) {
        a[i]     = (_Float16)p[i];
        a[8 + i] = (_Float16)p[8 + i];
      }
    }
    v16h bf;
    {
      const _Float16* p = ob + (size_t)(kb + hg * 16) * NTOK + tn * 16 + l15;
#pragma unroll
      for (int i = 0; i < 16; ++i) bf[i] = p[(size_t)i * NTOK];
    }
    acc = wmma_f16(a, bf, acc);
  }
  int token = tn * 16 + l15;
#pragma unroll
  for (int r = 0; r < 8; ++r) {
    int o = to * 16 + r + 8 * hg;
    out[((size_t)b * C_ + o) * NTOK + token] = acc[r] + bo[o];
  }
}

// ---------------------------------------------------------------------------
extern "C" void kernel_launch(void* const* d_in, const int* in_sizes, int n_in,
                              void* d_out, int out_size, void* d_ws,
                              size_t ws_size, hipStream_t stream) {
  const float* x  = (const float*)d_in[0];
  const float* wq = (const float*)d_in[1];
  const float* wk = (const float*)d_in[2];
  const float* wv = (const float*)d_in[3];
  const float* wo = (const float*)d_in[4];
  const float* bo = (const float*)d_in[5];
  float* out = (float*)d_out;

  // Workspace layout (f16): Q | K | V^T | O  == 4 x 4 MB = 16 MB
  size_t per = (size_t)B_ * H_ * NTOK * D_;
  _Float16* qws  = (_Float16*)d_ws;
  _Float16* kws  = qws + per;
  _Float16* vtws = kws + per;
  _Float16* ows  = vtws + per;

  dim3 gA(B_ * 8 * 256, 3);
  proj_qkv<<<gA, 32, 0, stream>>>(x, wq, wk, wv, qws, kws, vtws);
  flash_attn<<<B_ * H_ * 128, 32, 0, stream>>>(qws, kws, vtws, ows);
  proj_out<<<B_ * 8 * 256, 32, 0, stream>>>(ows, wo, bo, out);
}
